// SteeredConv2d_31877247271156
// MI455X (gfx1250) — matrix operand order repaired and checked by reading, not compiled, not run on hardware
//
#include <hip/hip_runtime.h>
#include <hip/hip_bf16.h>
#include <stdint.h>

// ---------------------------------------------------------------------------
// Steered 5x5 conv, fused as one implicit-GEMM with WMMA bf16x3 (hi/lo split)
//   M = 704 folded channels (f0 | fr k=0..3 | fi k=0..3 | u | v)
//   K = 25 taps x 64 in-channels, N = pixels
// Workspace layout (requires ~40.2 MB):
//   [0)                Xhi  : bf16 [8][132][132][64]  (NHWC, zero-padded)
//   [+17842176)        Xlo  : bf16 same
//   [+2*17842176)      Whi  : bf16 [44][25][2][32][16] (WMMA-A-swizzled)
//   [+...+2252800)     Wlo  : bf16 same
// ---------------------------------------------------------------------------

typedef __attribute__((ext_vector_type(16))) __bf16 v16bf;
typedef __attribute__((ext_vector_type(8)))  float  v8f;
typedef __attribute__((ext_vector_type(4)))  unsigned int v4u;
typedef __attribute__((ext_vector_type(8)))  int v8i_;
typedef __attribute__((ext_vector_type(4)))  int v4i_;

#define NB    8
#define NIN   64
#define NOUT  64
#define H     128
#define W     128
#define HP    132
#define WP    132
#define M_TOT 704
#define KTAP  25
#define KDIM  (KTAP*NIN)      // 1600
#define MT    44              // M tiles of 16
#define BLK_R 8               // pixel-block rows per WG (1 row per wave)
#define BLK_C 16              // pixel-block cols per WG (WMMA N)
#define PAT_R 12              // 8 + 4 halo
#define PAT_C 20              // 16 + 4 halo

union FragBF { v16bf v; uint4 q[2]; };

// ---------------------------------------------------------------------------
// basis helpers (closed-form replication of the python _make_basis)
// ---------------------------------------------------------------------------
__device__ __forceinline__ void tap_geom(int tap, float& r, float& th) {
    float cy = (float)(tap / 5 - 2);
    float cx = (float)(tap % 5 - 2);
    r  = sqrtf(cx * cx + cy * cy);
    th = atan2f(cy, cx);
}
__device__ __forceinline__ float ringf(float r, float r0) {
    float d = r - r0;
    return __expf(-d * d * 2.0f);           // sigma = 0.5 -> 1/(2s^2) = 2
}
__device__ float basis_b0(int b, int tap) { // normalized ring b = 0,1,2
    float n = 0.f;
    for (int t = 0; t < 25; ++t) { float r, th; tap_geom(t, r, th);
        float g = ringf(r, (float)b); n += g * g; }
    float r, th; tap_geom(tap, r, th);
    return ringf(r, (float)b) * rsqrtf(n);
}
// normalized g*cos(k*th) (isSin=0) or g*sin(k*th) (isSin=1); kidx=0..3 -> k=1..4
__device__ float basis_c(int kidx, int ridx, int isSin, int tap) {
    float kk = (float)(kidx + 1), r0 = (float)(ridx + 1);
    float n = 0.f;
    for (int t = 0; t < 25; ++t) {
        float r, th; tap_geom(t, r, th);
        float f = ringf(r, r0) * (isSin ? __sinf(kk * th) : __cosf(kk * th));
        n += f * f;
    }
    float r, th; tap_geom(tap, r, th);
    float f = ringf(r, r0) * (isSin ? __sinf(kk * th) : __cosf(kk * th));
    return f * rsqrtf(n);
}

// ---------------------------------------------------------------------------
// Kernel 0: pad x to 132x132, transpose NCHW->NHWC, split f32 -> bf16 hi+lo
// ---------------------------------------------------------------------------
__global__ void pad_split_x(const float* __restrict__ x,
                            __bf16* __restrict__ Xhi, __bf16* __restrict__ Xlo) {
    int idx = blockIdx.x * blockDim.x + threadIdx.x;
    if (idx >= NB * HP * WP * NIN) return;
    int i  = idx & 63;
    int xp = (idx >> 6) % WP;
    int yp = ((idx >> 6) / WP) % HP;
    int n  = idx / (64 * WP * HP);
    float v = 0.f;
    if (yp >= 2 && yp < HP - 2 && xp >= 2 && xp < WP - 2)
        v = x[(((size_t)n * NIN + i) * H + (yp - 2)) * W + (xp - 2)];
    __bf16 h = (__bf16)v;
    __bf16 l = (__bf16)(v - (float)h);
    Xhi[idx] = h;  Xlo[idx] = l;
}

// ---------------------------------------------------------------------------
// Kernel 1: fold basis into weights, emit WMMA-A-fragment-swizzled hi/lo.
// Swizzle: frag (mt,tap,cb) is 512 contiguous bf16: [lane 0..31][e 0..15]
//   row  m  = mt*16 + (lane&15)
//   chan i  = cb*32 + (lane>>4)*16 + e   (contiguous K per lane half)
// ---------------------------------------------------------------------------
__global__ void prep_weights(const float* __restrict__ wts,
                             const float* __restrict__ attw,
                             __bf16* __restrict__ Whi, __bf16* __restrict__ Wlo) {
    int idx = blockIdx.x * blockDim.x + threadIdx.x;
    if (idx >= M_TOT * KDIM) return;
    int e    = idx & 15;
    int lane = (idx >> 4) & 31;
    int blk  = idx >> 9;
    int cb   = blk & 1;
    int tap  = (blk >> 1) % 25;
    int mt   = blk / 50;
    int m    = mt * 16 + (lane & 15);
    int i    = cb * 32 + (lane >> 4) * 16 + e;

    float val = 0.f;
    if (m < 64) {                                     // f0 rows
        const float* wr = wts + ((size_t)m * NIN + i) * 19;
        for (int b = 0; b < 3; ++b) val += wr[b] * basis_b0(b, tap);
    } else if (m < 320) {                             // fr rows: wr*BR - wi*BI
        int k = (m - 64) >> 6, o = (m - 64) & 63;
        const float* wr = wts + ((size_t)o * NIN + i) * 19;
        for (int r = 0; r < 2; ++r)
            val += wr[3 + k * 2 + r]  * basis_c(k, r, 0, tap)
                 - wr[11 + k * 2 + r] * basis_c(k, r, 1, tap);
    } else if (m < 576) {                             // fi rows: wr*BI + wi*BR
        int k = (m - 320) >> 6, o = (m - 320) & 63;
        const float* wr = wts + ((size_t)o * NIN + i) * 19;
        for (int r = 0; r < 2; ++r)
            val += wr[3 + k * 2 + r]  * basis_c(k, r, 1, tap)
                 + wr[11 + k * 2 + r] * basis_c(k, r, 0, tap);
    } else if (m < 640) {                             // u rows: att * AR (k=1)
        int o = m - 576;
        const float* ar = attw + ((size_t)o * NIN + i) * 2;
        for (int r = 0; r < 2; ++r) val += ar[r] * basis_c(0, r, 0, tap);
    } else {                                          // v rows: att * AI (k=1)
        int o = m - 640;
        const float* ar = attw + ((size_t)o * NIN + i) * 2;
        for (int r = 0; r < 2; ++r) val += ar[r] * basis_c(0, r, 1, tap);
    }
    __bf16 h = (__bf16)val;
    __bf16 l = (__bf16)(val - (float)h);
    Whi[idx] = h;  Wlo[idx] = l;
}

// ---------------------------------------------------------------------------
// Kernel 2: fused conv-GEMM + steering. 8 waves; wave t owns pixel row y0+t
// (16 px) and all 44 M-tiles (352 accum VGPRs). TDM stages the halo patch.
// ---------------------------------------------------------------------------
#if defined(__has_builtin)
#if __has_builtin(__builtin_amdgcn_tensor_load_to_lds) && \
    __has_builtin(__builtin_amdgcn_s_wait_tensorcnt)
#define USE_TDM 1
#endif
#endif

__global__ __launch_bounds__(256, 1)
void steered_wmma(const __bf16* __restrict__ Xhi, const __bf16* __restrict__ Xlo,
                  const __bf16* __restrict__ Whi, const __bf16* __restrict__ Wlo,
                  const float* __restrict__ bias, float* __restrict__ out) {
    __shared__ __bf16 sHi[PAT_R * PAT_C * NIN];   // 30720 B
    __shared__ __bf16 sLo[PAT_R * PAT_C * NIN];   // 30720 B

    const int wg  = blockIdx.x;
    const int bx  = wg & 7;            // 8 col blocks of 16
    const int by  = (wg >> 3) & 15;    // 16 row blocks of 8
    const int n   = wg >> 7;           // batch
    const int tid  = threadIdx.x;
    const int lane = tid & 31;
    const int wv   = tid >> 5;         // wave id = pixel row in block
    const int y0 = by * BLK_R, x0 = bx * BLK_C;
    const int khalf = lane >> 4;
    const int ncol  = lane & 15;

#ifdef USE_TDM
    // ---- Tensor Data Mover: one 3D tile load per plane (X=64ch, Y=20, Z=12)
    // This toolchain carries the 6-arg builtin:
    //   (uint32x4 g0, int32x8 g1, int32x4 g2, int32x4 g3, int32x8 pad, i32 cpol)
    if (tid < 32) {
        const size_t gelem = (((size_t)n * HP + y0) * WP + x0) * NIN;
        unsigned long long gaHi = (unsigned long long)(uintptr_t)(Xhi + gelem);
        unsigned long long gaLo = (unsigned long long)(uintptr_t)(Xlo + gelem);
        unsigned ldsHi = (unsigned)(uintptr_t)(void*)sHi;  // low32 = LDS offset
        unsigned ldsLo = (unsigned)(uintptr_t)(void*)sLo;
        const unsigned s1 = (unsigned)(WP * NIN);          // 8448 elems
        v8i_ g1 = { (int)(1u << 16),                       // data_size=1 (2B)
                    (int)(64u << 16),                      // tensor_dim0=64
                    (int)((unsigned)WP << 16),             // tensor_dim1 lo16
                    (int)(64u << 16),                      // tile_dim0=64
                    (int)(PAT_C | (PAT_R << 16)),          // tile_dim1,2 = 20,12
                    (int)64,                               // dim0 stride
                    (int)((s1 & 0xFFFFu) << 16),           // dim1 stride lo16
                    (int)(s1 >> 16) };                     // dim1 stride hi32
        v4i_ g2 = { (int)HP, 0, 0, 0 };                    // tensor_dim2
        v4i_ g3 = { 0, 0, 0, 0 };
        v8i_ g4 = { 0, 0, 0, 0, 0, 0, 0, 0 };              // zero-filled extra group
        v4u g0h = { 1u, ldsHi, (unsigned)gaHi,
                    (unsigned)(gaHi >> 32) | (2u << 30) }; // count=1, type=2
        v4u g0l = { 1u, ldsLo, (unsigned)gaLo,
                    (unsigned)(gaLo >> 32) | (2u << 30) };
        __builtin_amdgcn_tensor_load_to_lds(g0h, g1, g2, g3, g4, 0);
        __builtin_amdgcn_tensor_load_to_lds(g0l, g1, g2, g3, g4, 0);
        __builtin_amdgcn_s_wait_tensorcnt(0);
    }
    __syncthreads();
#else
    for (int e = tid; e < PAT_R * PAT_C * NIN; e += 256) {
        int i  = e & 63;
        int pc = (e >> 6) % PAT_C;
        int pr = (e >> 6) / PAT_C;
        size_t g = (((size_t)n * HP + y0 + pr) * WP + (x0 + pc)) * NIN + i;
        sHi[e] = Xhi[g];  sLo[e] = Xlo[g];
    }
    __syncthreads();
#endif

    v8f acc[MT];
    #pragma unroll
    for (int t = 0; t < MT; ++t) acc[t] = (v8f)0.0f;

    // K loop: 25 taps x 2 channel-blocks; bf16x3 per tile pair
    for (int dy = 0; dy < 5; ++dy) {
        for (int dx = 0; dx < 5; ++dx) {
            const int tap  = dy * 5 + dx;
            const int prow = wv + dy;
            #pragma unroll
            for (int cb = 0; cb < 2; ++cb) {
                const int boff = ((prow * PAT_C) + dx + ncol) * NIN
                               + cb * 32 + khalf * 16;
                FragBF Bh, Bl;
                Bh.q[0] = *(const uint4*)(sHi + boff);
                Bh.q[1] = *(const uint4*)(sHi + boff + 8);
                Bl.q[0] = *(const uint4*)(sLo + boff);
                Bl.q[1] = *(const uint4*)(sLo + boff + 8);
                const size_t fbase0 = (((size_t)tap) * 2 + cb) * 512 + lane * 16;
                #pragma unroll
                for (int mt = 0; mt < MT; ++mt) {
                    const size_t fb = fbase0 + (size_t)mt * (KTAP * 2 * 512);
                    FragBF Ah, Al;
                    Ah.q[0] = *(const uint4*)(Whi + fb);
                    Ah.q[1] = *(const uint4*)(Whi + fb + 8);
                    Al.q[0] = *(const uint4*)(Wlo + fb);
                    Al.q[1] = *(const uint4*)(Wlo + fb + 8);
                    acc[mt] = __builtin_amdgcn_wmma_f32_16x16x32_bf16(
                        false, Ah.v, false, Bh.v, (short)0, acc[mt], false, false);
                    acc[mt] = __builtin_amdgcn_wmma_f32_16x16x32_bf16(
                        false, Ah.v, false, Bl.v, (short)0, acc[mt], false, false);
                    acc[mt] = __builtin_amdgcn_wmma_f32_16x16x32_bf16(
                        false, Al.v, false, Bh.v, (short)0, acc[mt], false, false);
                }
            }
        }
    }

    // ---- steering epilogue: pure elementwise on v8f accumulators ----------
    // C/D layout: element r of lane -> M = r + 8*khalf, N = ncol
    const int yy = y0 + wv;
    const int xx = x0 + ncol;
    #pragma unroll
    for (int ot = 0; ot < 4; ++ot) {
        v8f u = acc[36 + ot];
        v8f v = acc[40 + ot];
        v8f c1, s1;
        #pragma unroll
        for (int e = 0; e < 8; ++e) {
            float rho = sqrtf(u[e] * u[e] + v[e] * v[e] + 1e-8f);
            c1[e] = u[e] / rho;
            s1[e] = v[e] / rho;
        }
        v8f o8 = acc[ot];       // f0
        v8f ck = c1, sk = s1;
        #pragma unroll
        for (int k = 0; k < 4; ++k) {
            o8 += ck * acc[4 + 4 * k + ot] + sk * acc[20 + 4 * k + ot];
            v8f ck2 = ck * c1 - sk * s1;
            sk      = sk * c1 + ck * s1;
            ck      = ck2;
        }
        #pragma unroll
        for (int r = 0; r < 8; ++r) {
            int o = ot * 16 + r + khalf * 8;
            out[(((size_t)n * NOUT + o) * H + yy) * W + xx] = o8[r] + bias[o];
        }
    }
}

// ---------------------------------------------------------------------------
extern "C" void kernel_launch(void* const* d_in, const int* in_sizes, int n_in,
                              void* d_out, int out_size, void* d_ws, size_t ws_size,
                              hipStream_t stream) {
    const float* x    = (const float*)d_in[0];
    const float* wts  = (const float*)d_in[1];
    const float* attw = (const float*)d_in[2];
    const float* bias = (const float*)d_in[3];
    float* out = (float*)d_out;

    char* ws = (char*)d_ws;
    const size_t XB = (size_t)NB * HP * WP * NIN * 2;   // 17,842,176
    const size_t WB = (size_t)M_TOT * KDIM * 2;         //  2,252,800
    __bf16* Xhi = (__bf16*)(ws);
    __bf16* Xlo = (__bf16*)(ws + XB);
    __bf16* Whi = (__bf16*)(ws + 2 * XB);
    __bf16* Wlo = (__bf16*)(ws + 2 * XB + WB);

    {   // pad + NHWC transpose + bf16 hi/lo split
        int total = NB * HP * WP * NIN;
        pad_split_x<<<(total + 255) / 256, 256, 0, stream>>>(x, Xhi, Xlo);
    }
    {   // basis fold + WMMA-A swizzle
        int total = M_TOT * KDIM;
        prep_weights<<<(total + 255) / 256, 256, 0, stream>>>(wts, attw, Whi, Wlo);
    }
    {   // fused WMMA conv + steering: 8 batches x 16 row-blocks x 8 col-blocks
        steered_wmma<<<NB * 16 * 8, 256, 0, stream>>>(Xhi, Xlo, Whi, Wlo, bias, out);
    }
}
